// Modality_attention_89000312307943
// MI455X (gfx1250) — matrix operand order repaired and checked by reading, not compiled, not run on hardware
//
#include <hip/hip_runtime.h>
#include <hip/hip_bf16.h>
#include <stdint.h>

typedef __attribute__((ext_vector_type(16))) __bf16 v16bf;
typedef __attribute__((ext_vector_type(8)))  float  v8f;
typedef __attribute__((ext_vector_type(4)))  float  f32x4;

#define BATCH  8
#define HEADS  8
#define HDIM   64
#define INDIM  512
#define NSEQ   1024
#define SCALE  0.125f   // 64^-0.5

__device__ __forceinline__ __bf16 f2bf(float f) { return (__bf16)f; }  // v_cvt_pk_bf16_f32

// ---------------------------------------------------------------------------
// Prep: W [512][1024] f32 -> Wt [1024][512] bf16  (blockIdx.y selects matrix)
// ---------------------------------------------------------------------------
__global__ void prep_w_kernel(const float* __restrict__ W0, const float* __restrict__ W1,
                              __bf16* __restrict__ T0, __bf16* __restrict__ T1) {
  const float* W = blockIdx.y ? W1 : W0;
  __bf16*      T = blockIdx.y ? T1 : T0;
  int idx = blockIdx.x * blockDim.x + threadIdx.x;   // 0 .. 1024*512-1
  int n = idx >> 9;
  int k = idx & 511;
  T[idx] = f2bf(W[k * 1024 + n]);
}

// ---------------------------------------------------------------------------
// Prep: X [b][n][512] f32 -> Vt [b][h][d][n] bf16  (transposed values)
// ---------------------------------------------------------------------------
__global__ void prep_v_kernel(const float* __restrict__ X0, const float* __restrict__ X1,
                              __bf16* __restrict__ T0, __bf16* __restrict__ T1) {
  const float* X = blockIdx.y ? X1 : X0;
  __bf16*      T = blockIdx.y ? T1 : T0;
  int idx = blockIdx.x * blockDim.x + threadIdx.x;   // 0 .. 2^22-1
  int n = idx & 1023;
  int d = (idx >> 10) & 63;
  int h = (idx >> 16) & 7;
  int b = idx >> 19;
  T[idx] = f2bf(X[((size_t)(b * 1024 + n)) * 512 + h * 64 + d]);
}

// ---------------------------------------------------------------------------
// Projection GEMM: C[8192,1024] = A[8192,512] * W[512,1024] + bias
// 8 waves/block; wave -> 16-row stripe, 64 cols (4 WMMA col tiles).
// Ping-pong double buffering, fully unrolled (ks += 64): buffer swap is a
// register-allocation rename, not a mov chain.
// ---------------------------------------------------------------------------
__global__ __launch_bounds__(256)
void proj_kernel(const float* __restrict__ A0, const float* __restrict__ A1,
                 const __bf16* __restrict__ Wt0, const __bf16* __restrict__ Wt1,
                 const float* __restrict__ bias0, const float* __restrict__ bias1,
                 __bf16* __restrict__ q0, __bf16* __restrict__ q1,
                 __bf16* __restrict__ k0, __bf16* __restrict__ k1) {
  int z = blockIdx.z;
  const float*  A    = z ? A1 : A0;
  const __bf16* Wt   = z ? Wt1 : Wt0;
  const float*  bias = z ? bias1 : bias0;
  __bf16* qo = z ? q1 : q0;
  __bf16* ko = z ? k1 : k0;

  int wave  = threadIdx.x >> 5;
  int lane  = threadIdx.x & 31;
  int lhalf = lane >> 4;
  int lmod  = lane & 15;

  int row0 = blockIdx.y * 128 + wave * 16;  // M tile (b*n rows)
  int col0 = blockIdx.x * 64;               // N tile

  const float* Arow = A + (size_t)(row0 + lmod) * 512;

  auto loadA = [&](int ks) {
    int kbb = ks + lhalf * 16;
    f32x4 a0 = *(const f32x4*)(Arow + kbb);
    f32x4 a1 = *(const f32x4*)(Arow + kbb + 4);
    f32x4 a2 = *(const f32x4*)(Arow + kbb + 8);
    f32x4 a3 = *(const f32x4*)(Arow + kbb + 12);
    v16bf af;
#pragma unroll
    for (int j = 0; j < 4; ++j) {
      af[j]      = f2bf(a0[j]);
      af[4 + j]  = f2bf(a1[j]);
      af[8 + j]  = f2bf(a2[j]);
      af[12 + j] = f2bf(a3[j]);
    }
    return af;
  };
  auto loadB = [&](int ks, int t) {
    const __bf16* wp = Wt + (size_t)(col0 + t * 16 + lmod) * 512 + ks + lhalf * 16;
    return *(const v16bf*)wp;
  };

  v8f acc[4] = {};
  v16bf afA = loadA(0);
  v16bf bA[4];
#pragma unroll
  for (int t = 0; t < 4; ++t) bA[t] = loadB(0, t);

#pragma unroll
  for (int ks = 0; ks < 512; ks += 64) {
    // prefetch step ks+32 into the B buffers
    v16bf afB = loadA(ks + 32);
    v16bf bB[4];
#pragma unroll
    for (int t = 0; t < 4; ++t) bB[t] = loadB(ks + 32, t);

    // consume step ks (A buffers)
#pragma unroll
    for (int t = 0; t < 4; ++t)
      acc[t] = __builtin_amdgcn_wmma_f32_16x16x32_bf16(
          false, afA, false, bA[t], (short)0, acc[t], false, false);

    // prefetch step ks+64 into the A buffers (compile-time guarded)
    if (ks + 64 < 512) {
      afA = loadA(ks + 64);
#pragma unroll
      for (int t = 0; t < 4; ++t) bA[t] = loadB(ks + 64, t);
    }

    // consume step ks+32 (B buffers)
#pragma unroll
    for (int t = 0; t < 4; ++t)
      acc[t] = __builtin_amdgcn_wmma_f32_16x16x32_bf16(
          false, afB, false, bB[t], (short)0, acc[t], false, false);
  }

  // epilogue: bias + bf16 + scatter to q/k [b][h][n][64]
#pragma unroll
  for (int t = 0; t < 4; ++t) {
    int coln = col0 + t * 16 + lmod;
    float bv = bias[coln];
    __bf16* dst;
    int h, d;
    if (coln < 512) { h = coln >> 6; d = coln & 63; dst = qo; }
    else            { int c = coln - 512; h = c >> 6; d = c & 63; dst = ko; }
#pragma unroll
    for (int r = 0; r < 8; ++r) {
      int m    = r + lhalf * 8;          // C layout row
      int grow = row0 + m;
      int b = grow >> 10, n = grow & 1023;
      dst[(((size_t)(b * 8 + h)) * 1024 + n) * 64 + d] = f2bf(acc[t][r] + bv);
    }
  }
}

// ---------------------------------------------------------------------------
// Fused attention. Block = (b, 16-row n-tile, pair). wave == head.
// Softmax(dim=heads) is pointwise per (n,m) -> stream over 32-wide m tiles.
// All global loads of an m-step are issued up front; the softmax phase
// (barriers + LDS + v_exp_f32) covers the V-load latency.
// ---------------------------------------------------------------------------
__global__ __launch_bounds__(256)
void attn_kernel(const __bf16* __restrict__ QN, const __bf16* __restrict__ KN,
                 const __bf16* __restrict__ QR, const __bf16* __restrict__ KR,
                 const __bf16* __restrict__ VTN, const __bf16* __restrict__ VTR,
                 float* __restrict__ out) {
  __shared__ float s_sc[8][16][33];   // scores, then exp'd scores, per head
  __shared__ float s_rd[16][33];      // 1 / sum_h(exp)

  int pair = blockIdx.z;
  const __bf16* Q  = pair ? QR  : QN;   // attn1: noise_q . rgb_k ; attn2: rgb_q . noise_k
  const __bf16* K  = pair ? KN  : KR;
  const __bf16* Vt = pair ? VTN : VTR;  // noise out uses rv, rgb out uses nv
  float* O = out + (size_t)pair * ((size_t)BATCH * NSEQ * INDIM);

  int b  = blockIdx.y;
  int n0 = blockIdx.x * 16;
  int h  = threadIdx.x >> 5;
  int lane  = threadIdx.x & 31;
  int lhalf = lane >> 4;
  int lmod  = lane & 15;
  int kb    = lhalf * 8;

  // Q fragments (A layout), K dim = head dim 64 -> two 16x32 fragments
  const __bf16* qrow = Q + (((size_t)(b * 8 + h)) * 1024 + n0 + lmod) * 64;
  v16bf qf0, qf1;
#pragma unroll
  for (int j = 0; j < 8; ++j) {
    qf0[j]     = qrow[2 * kb + j];
    qf0[8 + j] = qrow[2 * kb + 8 + j];
    qf1[j]     = qrow[32 + 2 * kb + j];
    qf1[8 + j] = qrow[40 + 2 * kb + j];
  }

  const __bf16* Kbase = K  + ((size_t)(b * 8 + h)) * 1024 * 64;  // [m][d]
  const __bf16* Vbase = Vt + ((size_t)(b * 8 + h)) * 64 * 1024;  // [d][m]

  v8f oacc[4] = {};   // 16n x 64d output accumulator

  for (int ms = 0; ms < NSEQ; ms += 32) {
    // ---- issue ALL global loads for this m-step up front ----
    v16bf kf[4];   // [mt*2 + chunk]: chunk 0 = d0..31, chunk 1 = d32..63
#pragma unroll
    for (int mt = 0; mt < 2; ++mt) {
      const __bf16* krow = Kbase + (size_t)(ms + mt * 16 + lmod) * 64;
      kf[mt * 2 + 0] = *(const v16bf*)(krow + lhalf * 16);
      kf[mt * 2 + 1] = *(const v16bf*)(krow + 32 + lhalf * 16);
    }
    v16bf vf[4];   // B fragments of V for the 4 d-col tiles (needed after softmax)
#pragma unroll
    for (int dt = 0; dt < 4; ++dt) {
      const __bf16* vrow = Vbase + (size_t)(dt * 16 + lmod) * 1024 + ms + lhalf * 16;
      vf[dt] = *(const v16bf*)vrow;
    }

    // ---- scores for this head: two 16x16 tiles over m ----
#pragma unroll
    for (int mt = 0; mt < 2; ++mt) {
      v8f c = {};
      c = __builtin_amdgcn_wmma_f32_16x16x32_bf16(false, qf0, false, kf[mt * 2 + 0],
                                                  (short)0, c, false, false);
      c = __builtin_amdgcn_wmma_f32_16x16x32_bf16(false, qf1, false, kf[mt * 2 + 1],
                                                  (short)0, c, false, false);
#pragma unroll
      for (int r = 0; r < 8; ++r)
        s_sc[h][r + lhalf * 8][mt * 16 + lmod] = c[r] * SCALE;
    }
    __syncthreads();

    // ---- cross-head softmax: each thread owns 2 (n,m) columns; exp stored
    //      back in place so the P-build needs no transcendentals ----
    for (int p = threadIdx.x; p < 512; p += 256) {
      int n = p >> 5, m = p & 31;
      float mx = s_sc[0][n][m];
#pragma unroll
      for (int hh = 1; hh < 8; ++hh) mx = fmaxf(mx, s_sc[hh][n][m]);
      float s = 0.f;
#pragma unroll
      for (int hh = 0; hh < 8; ++hh) {
        float e = __expf(s_sc[hh][n][m] - mx);
        s_sc[hh][n][m] = e;
        s += e;
      }
      s_rd[n][m] = 1.0f / s;
    }
    __syncthreads();

    // ---- P as bf16 A fragment (16n x 32m): pure LDS read + multiply ----
    v16bf pf;
#pragma unroll
    for (int j = 0; j < 8; ++j) {
      int m1 = 2 * kb + j;
      int m2 = 2 * kb + 8 + j;
      pf[j]     = f2bf(s_sc[h][lmod][m1] * s_rd[lmod][m1]);
      pf[8 + j] = f2bf(s_sc[h][lmod][m2] * s_rd[lmod][m2]);
    }

    // ---- oacc += P[16x32] * V[32x64] ----
#pragma unroll
    for (int dt = 0; dt < 4; ++dt)
      oacc[dt] = __builtin_amdgcn_wmma_f32_16x16x32_bf16(
          false, pf, false, vf[dt], (short)0, oacc[dt], false, false);
    __syncthreads();   // protect s_sc before next iteration rewrites it
  }

  // ---- write merged-heads fp32 output [b][n][h*64+d] ----
#pragma unroll
  for (int dt = 0; dt < 4; ++dt) {
    int d = dt * 16 + lmod;
#pragma unroll
    for (int r = 0; r < 8; ++r) {
      int n = n0 + r + lhalf * 8;
      O[((size_t)(b * 1024 + n)) * 512 + h * 64 + d] = oacc[dt][r];
    }
  }
}

// ---------------------------------------------------------------------------
extern "C" void kernel_launch(void* const* d_in, const int* in_sizes, int n_in,
                              void* d_out, int out_size, void* d_ws, size_t ws_size,
                              hipStream_t stream) {
  const float* nv = (const float*)d_in[0];
  const float* rv = (const float*)d_in[1];
  const float* Wn = (const float*)d_in[2];
  const float* bn = (const float*)d_in[3];
  const float* Wr = (const float*)d_in[4];
  const float* br = (const float*)d_in[5];
  float* out = (float*)d_out;

  char* ws = (char*)d_ws;
  const size_t qk_bytes = (size_t)BATCH * HEADS * NSEQ * HDIM * sizeof(__bf16); // 8 MiB
  __bf16* QN  = (__bf16*)(ws + 0 * qk_bytes);
  __bf16* KN  = (__bf16*)(ws + 1 * qk_bytes);
  __bf16* QR  = (__bf16*)(ws + 2 * qk_bytes);
  __bf16* KR  = (__bf16*)(ws + 3 * qk_bytes);
  __bf16* VTN = (__bf16*)(ws + 4 * qk_bytes);
  __bf16* VTR = (__bf16*)(ws + 5 * qk_bytes);
  __bf16* WTN = (__bf16*)(ws + 6 * qk_bytes);
  __bf16* WTR = (__bf16*)(ws + 6 * qk_bytes + (size_t)(1024 * 512 * sizeof(__bf16)));

  prep_w_kernel<<<dim3(2048, 2), 256, 0, stream>>>(Wn, Wr, WTN, WTR);
  prep_v_kernel<<<dim3(16384, 2), 256, 0, stream>>>(nv, rv, VTN, VTR);
  proj_kernel<<<dim3(16, 64, 2), 256, 0, stream>>>(nv, rv, WTN, WTR, bn, br,
                                                   QN, QR, KN, KR);
  attn_kernel<<<dim3(64, 8, 2), 256, 0, stream>>>(QN, KN, QR, KR, VTN, VTR, out);
}
